// PointerLSTM_48911087566944
// MI455X (gfx1250) — matrix operand order repaired and checked by reading, not compiled, not run on hardware
//
#include <hip/hip_runtime.h>
#include <hip/hip_bf16.h>

// ---------------------------------------------------------------------------
// PointerLSTM for MI455X (gfx1250, wave32, WMMA).
// convT(x4) -> gemm_w2e (LDS-staged WMMA f16, NT x stream) ->
// lstm_seq (WMMA, 1 WG, cell state in regs) ->
// attn (v_tanh_f32 + shfl dot + LDS softmax over L2-resident f16 w2_e).
// ---------------------------------------------------------------------------

typedef __attribute__((ext_vector_type(16))) _Float16 v16h;
typedef __attribute__((ext_vector_type(8)))  _Float16 v8h;
typedef __attribute__((ext_vector_type(8)))  float    v8f;
typedef __attribute__((ext_vector_type(4)))  float    v4f;

union H16 { v16h v; v8h h[2]; _Float16 e[16]; };

// ---- CDNA5 WMMA helpers (layouts per CDNA5 ISA §7.12.2) ---------

__device__ __forceinline__ v8f wmma16(v16h a, v16h b, v8f c) {
  // D = A(16x32 f16) x B(32x16 f16) + C(16x16 f32)
  return __builtin_amdgcn_wmma_f32_16x16x32_f16(
      /*neg_a=*/false, a, /*neg_b=*/false, b,
      /*c_mod=*/(short)0, c, /*reuse_a=*/false, /*reuse_b=*/false);
}

// A-frag from LDS f16 tile with row stride 264 (padded vs 64 banks).
// Lane L: m=L&15, half=L>>4. VGPR j=0..3 <- K = 32*kt + 8*half + {0..7},
//                            VGPR j=4..7 <- K = 32*kt + 16 + 8*half + {0..7}.
__device__ __forceinline__ v16h load_a_lds(const _Float16* base, int lane, int kt) {
  int m = lane & 15, hf = lane >> 4;
  const _Float16* p = base + m * 264 + kt * 32 + hf * 8;
  H16 u;
  u.h[0] = *(const v8h*)(p);        // 16B aligned (264*2 = 33*16)
  u.h[1] = *(const v8h*)(p + 16);
  return u.v;
}

// A-frag from global fp32 (row stride 256 floats), nontemporal (stream-once).
__device__ __forceinline__ v16h load_a_glb_f32_nt(const float* X, long row0,
                                                  int lane, int kt) {
  int m = lane & 15, hf = lane >> 4;
  const float* p = X + (row0 + m) * 256 + kt * 32 + hf * 8;
  v4f a0 = __builtin_nontemporal_load((const v4f*)p);
  v4f a1 = __builtin_nontemporal_load((const v4f*)p + 1);
  v4f a2 = __builtin_nontemporal_load((const v4f*)(p + 16));
  v4f a3 = __builtin_nontemporal_load((const v4f*)(p + 16) + 1);
  H16 u;
#pragma unroll
  for (int i = 0; i < 4; ++i) { u.e[i]      = (_Float16)a0[i];
                                u.e[4 + i]  = (_Float16)a1[i];
                                u.e[8 + i]  = (_Float16)a2[i];
                                u.e[12 + i] = (_Float16)a3[i]; }
  return u.v;
}

// B-frag from global f16 matrix stored transposed [N][K=256].
// Lane L: n = nBase + (L&15); K = 32*kt + 8*(L>>4) + {0..7}, then 32*kt + 16 + 8*(L>>4) + {0..7}.
__device__ __forceinline__ v16h load_b_glb(const _Float16* Bt, int nBase, int lane, int kt) {
  int n = nBase + (lane & 15), hf = lane >> 4;
  const _Float16* p = Bt + (long)n * 256 + kt * 32 + hf * 8;
  H16 u;
  u.h[0] = *(const v8h*)(p);
  u.h[1] = *(const v8h*)(p + 16);
  return u.v;
}

// B-frag from LDS panel [64 rows][264 padded], row = local n.
__device__ __forceinline__ v16h load_b_lds(const _Float16* sB, int nBase, int lane, int kt) {
  int n = nBase + (lane & 15), hf = lane >> 4;
  const _Float16* p = sB + n * 264 + kt * 32 + hf * 8;
  H16 u;
  u.h[0] = *(const v8h*)(p);
  u.h[1] = *(const v8h*)(p + 16);
  return u.v;
}

// C/D frag: VGPR r, lane L -> row = row0 + r + 8*(L>>4), col = n0 + (L&15).
__device__ __forceinline__ void store_c_f16(_Float16* C, long row0, int ldc, int n0,
                                            int lane, v8f acc) {
  int col = n0 + (lane & 15), hf = lane >> 4;
#pragma unroll
  for (int r = 0; r < 8; ++r)
    C[(row0 + r + 8 * hf) * (long)ldc + col] = (_Float16)acc[r];
}

__device__ __forceinline__ void store_c_f32(float* C, long row0, int ldc, int n0,
                                            int lane, v8f acc) {
  int col = n0 + (lane & 15), hf = lane >> 4;
#pragma unroll
  for (int r = 0; r < 8; ++r)
    C[(row0 + r + 8 * hf) * (long)ldc + col] = acc[r];
}

// ---- transcendentals -------------------------------------------------------

__device__ __forceinline__ float fast_tanh(float x) {
  float r;
  asm volatile("v_tanh_f32 %0, %1" : "=v"(r) : "v"(x));   // gfx1250 TRANS op
  return r;
}
__device__ __forceinline__ float sigm(float x) { return 1.f / (1.f + __expf(-x)); }

// ---------------------------------------------------------------------------
// Kernel T: fp32 (K,N) row-major -> f16 transposed [N][K]
// ---------------------------------------------------------------------------
__global__ void __launch_bounds__(256) convT(const float* __restrict__ src,
                                             _Float16* __restrict__ dst,
                                             int K, int N) {
  int i = blockIdx.x * 256 + threadIdx.x;
  if (i < K * N) {
    int k = i / N, n = i - k * N;
    dst[(long)n * K + k] = (_Float16)src[i];
  }
}

// ---------------------------------------------------------------------------
// Kernel B: w2e = x @ W2, f16 out (64MB -> L2-resident for the 16 attn passes).
// 1024 blocks x 8 waves; wave owns a 16-row strip and all 16 N-tiles.
// All 8 A-frags live in registers (x read exactly once, NT); W2t staged in
// LDS in four 64-row panels; per-kt B-frags preloaded into regs so the
// ds loads clause ahead of the 4 WMMAs.
// ---------------------------------------------------------------------------
__global__ void __launch_bounds__(256) gemm_w2e(const float* __restrict__ x,
                                                const _Float16* __restrict__ W2t,
                                                _Float16* __restrict__ w2e) {
  __shared__ _Float16 sB[64 * 264];           // 33.8 KB panel
  const int tid = threadIdx.x, wave = tid >> 5, lane = tid & 31;
  const long row0 = (long)blockIdx.x * 128 + wave * 16;

  v16h aR[8];
#pragma unroll
  for (int kt = 0; kt < 8; ++kt) aR[kt] = load_a_glb_f32_nt(x, row0, lane, kt);

  v8f acc[16];
#pragma unroll
  for (int nt = 0; nt < 16; ++nt) acc[nt] = (v8f)(0.f);

#pragma unroll
  for (int p = 0; p < 4; ++p) {               // N panels of 64 columns
    __syncthreads();                          // previous panel fully consumed
    for (int i = tid; i < 2048; i += 256) {   // 64 rows x 32 v8h chunks
      int r = i >> 5, ch = i & 31;
      *(v8h*)(sB + r * 264 + ch * 8) =
          *(const v8h*)(W2t + (long)(64 * p + r) * 256 + ch * 8);
    }
    __syncthreads();
#pragma unroll
    for (int kt = 0; kt < 8; ++kt) {
      v16h bar[4];
#pragma unroll
      for (int q = 0; q < 4; ++q) bar[q] = load_b_lds(sB, q * 16, lane, kt);
#pragma unroll
      for (int q = 0; q < 4; ++q)
        acc[4 * p + q] = wmma16(aR[kt], bar[q], acc[4 * p + q]);
    }
  }
#pragma unroll
  for (int nt = 0; nt < 16; ++nt) store_c_f16(w2e, row0, 256, nt * 16, lane, acc[nt]);
}

// ---------------------------------------------------------------------------
// Kernel C: sequential LSTM, ONE workgroup (32 wave32s on one WGP).
// Wave w: mt = w&1 (16-row b tile), jt = w>>1 (16-col j tile).
// Gate mapping puts i/f/g/o for the same (b,j) cell into one thread's 4 accs,
// so the cell state c lives in 8 registers per thread across all 16 steps.
// ---------------------------------------------------------------------------
__global__ void __launch_bounds__(1024) lstm_seq(const float* __restrict__ x,
                                                 const _Float16* __restrict__ kT,
                                                 const _Float16* __restrict__ Rt,
                                                 const _Float16* __restrict__ W1t,
                                                 const float* __restrict__ bias,
                                                 float* __restrict__ w1d_all) {
  __shared__ _Float16 hA[32 * 264];          // h (or x_last) as f16, padded rows
  const int tid = threadIdx.x, lane = tid & 31, wave = tid >> 5;
  const int mt = wave & 1, jt = wave >> 1;   // jt in [0,16)
  const _Float16* aBase = hA + mt * 16 * 264;

  // stage x_last = x[:, T-1, :] as f16
  for (int i = tid; i < 32 * 256; i += 1024) {
    int b = i >> 8, d = i & 255;
    hA[b * 264 + d] = (_Float16)x[((long)b * 4096 + 4095) * 256 + d];
  }
  __syncthreads();

  // z_x = x_last @ kernel + bias  (kept in registers for all 16 steps)
  v8f zx[4];
#pragma unroll
  for (int g = 0; g < 4; ++g) zx[g] = (v8f)(0.f);
  for (int kt = 0; kt < 8; ++kt) {
    v16h a = load_a_lds(aBase, lane, kt);
    v16h br[4];
#pragma unroll
    for (int g = 0; g < 4; ++g) br[g] = load_b_glb(kT, 256 * g + 16 * jt, lane, kt);
#pragma unroll
    for (int g = 0; g < 4; ++g) zx[g] = wmma16(a, br[g], zx[g]);
  }
#pragma unroll
  for (int g = 0; g < 4; ++g) {
    float bv = bias[256 * g + 16 * jt + (lane & 15)];
#pragma unroll
    for (int r = 0; r < 8; ++r) zx[g][r] += bv;
  }
  __syncthreads();                            // zx reads of hA done

  // h0 = 0, c0 = 0
  for (int i = tid; i < 32 * 264; i += 1024) hA[i] = (_Float16)0.f;
  float c[8];
#pragma unroll
  for (int r = 0; r < 8; ++r) c[r] = 0.f;
  __syncthreads();

  const int colj = 16 * jt + (lane & 15);
  const int hf = lane >> 4;

  for (int s = 0; s < 16; ++s) {
    // z = z_x + h @ recurrent_kernel
    v8f z[4];
#pragma unroll
    for (int g = 0; g < 4; ++g) z[g] = zx[g];
    for (int kt = 0; kt < 8; ++kt) {
      v16h a = load_a_lds(aBase, lane, kt);
      v16h br[4];
#pragma unroll
      for (int g = 0; g < 4; ++g) br[g] = load_b_glb(Rt, 256 * g + 16 * jt, lane, kt);
#pragma unroll
      for (int g = 0; g < 4; ++g) z[g] = wmma16(a, br[g], z[g]);
    }
    __syncthreads();                          // all hA reads done before rewrite

    // gates -> cell update -> new h (f16) into hA
#pragma unroll
    for (int r = 0; r < 8; ++r) {
      float iv = sigm(z[0][r]);
      float fv = sigm(z[1][r]);
      float gv = fast_tanh(z[2][r]);
      float ov = sigm(z[3][r]);
      c[r] = fv * c[r] + iv * gv;
      float hv = ov * fast_tanh(c[r]);
      int b = 16 * mt + r + 8 * hf;
      hA[b * 264 + colj] = (_Float16)hv;
    }
    __syncthreads();

    // w1_d[s] = h @ W1  (wave w -> (mt, nt=jt) tile of the 32x256 result)
    v8f wd = (v8f)(0.f);
    for (int kt = 0; kt < 8; ++kt) {
      v16h a = load_a_lds(aBase, lane, kt);
      v16h b = load_b_glb(W1t, 16 * jt, lane, kt);
      wd = wmma16(a, b, wd);
    }
    store_c_f32(w1d_all + (long)s * 32 * 256, 16 * mt, 256, 16 * jt, lane, wd);
    __syncthreads();
  }
}

// ---------------------------------------------------------------------------
// Kernel D: all 16 attention passes in parallel. Block = one (s,b) pair.
// Lane l owns h-dims [8l, 8l+8); wave handles rows t = wave, wave+32, ...
// scores -> LDS -> softmax over T=4096 -> probs to out[b][s][t].
// ---------------------------------------------------------------------------
__global__ void __launch_bounds__(1024) attn(const _Float16* __restrict__ w2e,
                                             const float* __restrict__ w1d_all,
                                             const float* __restrict__ V,
                                             float* __restrict__ out) {
  __shared__ float sc[4096];
  __shared__ float red[32];
  const int tid = threadIdx.x, lane = tid & 31, wave = tid >> 5;
  const int s = blockIdx.x >> 5, b = blockIdx.x & 31;

  float w1r[8], vr[8];
  const float* wp = w1d_all + ((long)(s * 32 + b)) * 256 + lane * 8;
#pragma unroll
  for (int i = 0; i < 8; ++i) { w1r[i] = wp[i]; vr[i] = V[lane * 8 + i]; }

  const _Float16* E = w2e + (long)b * 4096 * 256;
  for (int t = wave; t < 4096; t += 32) {
    const _Float16* ep = E + (long)t * 256 + lane * 8;
    float p = 0.f;
#pragma unroll
    for (int i = 0; i < 8; ++i) p += fast_tanh(w1r[i] + (float)ep[i]) * vr[i];
#pragma unroll
    for (int off = 16; off > 0; off >>= 1) p += __shfl_xor(p, off);
    if (lane == 0) sc[t] = p;
  }
  __syncthreads();

  // max over T
  float mx = -3.4e38f;
  for (int t = tid; t < 4096; t += 1024) mx = fmaxf(mx, sc[t]);
#pragma unroll
  for (int off = 16; off > 0; off >>= 1) mx = fmaxf(mx, __shfl_xor(mx, off));
  if (lane == 0) red[wave] = mx;
  __syncthreads();
  if (wave == 0) {
    float m2 = red[lane];
#pragma unroll
    for (int off = 16; off > 0; off >>= 1) m2 = fmaxf(m2, __shfl_xor(m2, off));
    if (lane == 0) red[0] = m2;
  }
  __syncthreads();
  mx = red[0];

  // exp + sum
  float ssum = 0.f;
  for (int t = tid; t < 4096; t += 1024) {
    float e = __expf(sc[t] - mx);
    sc[t] = e;
    ssum += e;
  }
#pragma unroll
  for (int off = 16; off > 0; off >>= 1) ssum += __shfl_xor(ssum, off);
  __syncthreads();                       // red[0] consumed before reuse
  if (lane == 0) red[wave] = ssum;
  __syncthreads();
  if (wave == 0) {
    float s2 = red[lane];
#pragma unroll
    for (int off = 16; off > 0; off >>= 1) s2 += __shfl_xor(s2, off);
    if (lane == 0) red[0] = s2;
  }
  __syncthreads();
  float inv = 1.f / red[0];

  float* op = out + ((long)b * 16 + s) * 4096;
  for (int t = tid; t < 4096; t += 1024)
    __builtin_nontemporal_store(sc[t] * inv, op + t);   // write-once output
}

// ---------------------------------------------------------------------------

extern "C" void kernel_launch(void* const* d_in, const int* in_sizes, int n_in,
                              void* d_out, int out_size, void* d_ws, size_t ws_size,
                              hipStream_t stream) {
  (void)in_sizes; (void)n_in; (void)out_size; (void)ws_size;
  const float* x    = (const float*)d_in[0];   // (32,4096,256)
  const float* kern = (const float*)d_in[1];   // (256,1024)
  const float* rker = (const float*)d_in[2];   // (256,1024)
  const float* bias = (const float*)d_in[3];   // (1024,)
  const float* W1   = (const float*)d_in[4];   // (256,256)
  const float* W2   = (const float*)d_in[5];   // (256,256)
  const float* V    = (const float*)d_in[6];   // (256,1)
  float* out = (float*)d_out;                  // (32,16,4096)

  // workspace layout (~69 MB)
  char* ws = (char*)d_ws;
  _Float16* w2e = (_Float16*)(ws);                       // 131072*256*2 = 64 MiB
  _Float16* Rt  = (_Float16*)(ws + 67108864);            // 1024*256*2
  _Float16* kT  = (_Float16*)(ws + 67108864 + 524288);   // 1024*256*2
  _Float16* W2t = (_Float16*)(ws + 67108864 + 1048576);  // 256*256*2
  _Float16* W1t = (_Float16*)(ws + 67108864 + 1179648);  // 256*256*2
  float*    w1d = (float*)   (ws + 67108864 + 1310720);  // 16*32*256*4

  convT<<<(256 * 1024 + 255) / 256, 256, 0, stream>>>(kern, kT, 256, 1024);
  convT<<<(256 * 1024 + 255) / 256, 256, 0, stream>>>(rker, Rt, 256, 1024);
  convT<<<(256 * 256 + 255) / 256, 256, 0, stream>>>(W2, W2t, 256, 256);
  convT<<<(256 * 256 + 255) / 256, 256, 0, stream>>>(W1, W1t, 256, 256);

  gemm_w2e<<<1024, 256, 0, stream>>>(x, W2t, w2e);       // 17.2 GFLOP WMMA
  lstm_seq<<<1, 1024, 0, stream>>>(x, kT, Rt, W1t, bias, w1d);
  attn<<<512, 1024, 0, stream>>>(w2e, w1d, V, out);      // 16 L2-resident passes
}
